// Encoder_27900107554918
// MI455X (gfx1250) — matrix operand order repaired and checked by reading, not compiled, not run on hardware
//
#include <hip/hip_runtime.h>

typedef _Float16 v16h __attribute__((ext_vector_type(16)));
typedef float    v8f  __attribute__((ext_vector_type(8)));

#define Bn 512
#define Tn 1024
#define Hn 32
#define Ln 16

__device__ __forceinline__ float fast_tanh(float x) {
#if __has_builtin(__builtin_amdgcn_tanhf)
    return __builtin_amdgcn_tanhf(x);            // gfx1250 v_tanh_f32
#elif __has_builtin(__builtin_amdgcn_tanh_f32)
    return __builtin_amdgcn_tanh_f32(x);
#else
    float e = __expf(x + x);
    return 1.0f - 2.0f / (e + 1.0f);
#endif
}
__device__ __forceinline__ float fast_sig(float x) {
    return 0.5f * fast_tanh(0.5f * x) + 0.5f;    // one trans op per sigmoid
}

// Block = 64 threads = 2 waves per 16-batch-row tile.
// Wave wv owns gate N-tiles {wv, 2+wv, 4+wv, 6+wv} (= i,f,g,o for hidden cols 16*wv..16*wv+15)
// and produces h/c for hidden column group jg = wv. Waves exchange h via LDS.
//
// MODE 0: layer0  (f32 x [B,T,2] input, writes f16 h-seq)
// MODE 1: layer1  (f16 h-seq input,     writes f16 h-seq)
// MODE 2: layer2  (f16 h-seq input,     writes final f32 cell state)
template<int MODE>
__global__ void __launch_bounds__(64)
lstm_wave_kernel(const float* __restrict__ x0,
                 const unsigned short* __restrict__ xin_u16,
                 const float* __restrict__ W_ih,
                 const float* __restrict__ W_hh,
                 const float* __restrict__ b_ih,
                 const float* __restrict__ b_hh,
                 unsigned short* __restrict__ hout_u16,
                 float* __restrict__ c_out)
{
    constexpr int P = 40;                        // LDS row pitch (halfs): 80B rows, 16B-aligned chunks
    __shared__ __align__(16) _Float16 hbuf[16 * P];

    const int tid = threadIdx.x;
    const int wv  = tid >> 5;                    // wave id: owns column group jg = wv
    const int l   = tid & 31;                    // lane in wave
    const int n   = l & 15;                      // B/C/D column within tile
    const int hi  = l >> 4;                      // lane half-group
    const int b0  = blockIdx.x * 16;             // batch tile origin

    const _Float16* xin  = (const _Float16*)xin_u16;
    _Float16*       hout = (_Float16*)hout_u16;

    // ---- Resident B-fragments (f32 -> f16, once). B 32x16 layout: lane holds
    // K values of column n: K = hi*8 + (e&7) + (e>>3)*16.  nt = 2q + wv.
    const int kb = hi * 8;
    v16h Bhh[4];
#pragma unroll
    for (int q = 0; q < 4; ++q) {
        const float* wr = W_hh + (size_t)((2 * q + wv) * 16 + n) * Hn + kb;
#pragma unroll
        for (int e = 0; e < 16; ++e) Bhh[q][e] = (_Float16)wr[(e & 7) + (e >> 3) * 16];
    }

    v16h  Bih[4];
    float w0c[4], w1c[4];
    if constexpr (MODE == 0) {
#pragma unroll
        for (int q = 0; q < 4; ++q) {            // W_ih0 is [128,2]
            w0c[q] = W_ih[((2 * q + wv) * 16 + n) * 2 + 0];
            w1c[q] = W_ih[((2 * q + wv) * 16 + n) * 2 + 1];
        }
    } else {
#pragma unroll
        for (int q = 0; q < 4; ++q) {
            const float* wr = W_ih + (size_t)((2 * q + wv) * 16 + n) * Hn + kb;
#pragma unroll
            for (int e = 0; e < 16; ++e) Bih[q][e] = (_Float16)wr[(e & 7) + (e >> 3) * 16];
        }
    }

    float bsc[4];
#pragma unroll
    for (int q = 0; q < 4; ++q) {
        const int g = (2 * q + wv) * 16 + n;
        bsc[q] = b_ih[g] + b_hh[g];
    }

    // h0 = 0
    for (int i = tid; i < 16 * P; i += 64) hbuf[i] = (_Float16)0.0f;
    __syncthreads();

    // cell state (this wave's column half) in C/D layout: cs[r] = c[row r+8*hi][col 16*wv+n]
    v8f cs;
#pragma unroll
    for (int r = 0; r < 8; ++r) cs[r] = 0.0f;

    // A-fragment addressing (16-bit 16x32): lane row = l%16; K-chunks {0..7,16..23} / {8..15,24..31}
    const int mA  = n;
    const int k0a = hi * 8;
    const int k0b = k0a + 16;

    // ---- software-pipelined input prefetch (x_t has no recurrence dependence) ----
    uint4  pf0, pf1;                             // MODE 1/2: next x A-frag chunks
    float2 pfx[8];                               // MODE 0:   next x rows
    if constexpr (MODE == 0) {
#pragma unroll
        for (int r = 0; r < 8; ++r)
            pfx[r] = *(const float2*)(x0 + ((size_t)(b0 + hi * 8 + r) * Tn + 0) * 2);
    } else {
        const _Float16* xrow = xin + ((size_t)(b0 + mA) * Tn + 0) * Hn;
        pf0 = *(const uint4*)(xrow + k0a);
        pf1 = *(const uint4*)(xrow + k0b);
    }

    for (int t = 0; t < Tn; ++t) {
        // h_{t-1} A-fragment from LDS (2x ds_load_b128); both waves read the full row
        union { v16h v; uint4 q[2]; } ah;
        ah.q[0] = *(const uint4*)&hbuf[mA * P + k0a];
        ah.q[1] = *(const uint4*)&hbuf[mA * P + k0b];

        // consume prefetched x_t, then issue loads for x_{t+1} (latency hidden under WMMA+tanh)
        union { v16h v; uint4 q[2]; } ax;
        float2 xcur[8];
        const int tn = (t + 1 < Tn) ? t + 1 : t;
        if constexpr (MODE == 0) {
#pragma unroll
            for (int r = 0; r < 8; ++r) xcur[r] = pfx[r];
#pragma unroll
            for (int r = 0; r < 8; ++r)
                pfx[r] = *(const float2*)(x0 + ((size_t)(b0 + hi * 8 + r) * Tn + tn) * 2);
        } else {
            ax.q[0] = pf0; ax.q[1] = pf1;
            const _Float16* xrow = xin + ((size_t)(b0 + mA) * Tn + tn) * Hn;
            pf0 = *(const uint4*)(xrow + k0a);
            pf1 = *(const uint4*)(xrow + k0b);
        }

        // gate accumulators (this wave's 4 N-tiles), seeded with (b_ih + b_hh)
        v8f acc[4];
#pragma unroll
        for (int q = 0; q < 4; ++q)
#pragma unroll
            for (int r = 0; r < 8; ++r) acc[q][r] = bsc[q];

        if constexpr (MODE == 0) {
#pragma unroll
            for (int r = 0; r < 8; ++r)
#pragma unroll
                for (int q = 0; q < 4; ++q)
                    acc[q][r] = fmaf(w0c[q], xcur[r].x, fmaf(w1c[q], xcur[r].y, acc[q][r]));
        } else {
#pragma unroll
            for (int q = 0; q < 4; ++q)
                acc[q] = __builtin_amdgcn_wmma_f32_16x16x32_f16(
                    false, ax.v, false, Bih[q], (short)0, acc[q], false, false);
        }

        // all waves have read h_{t-1} from LDS; safe to overwrite after this barrier
        __syncthreads();

        // recurrent GEMM: 4 WMMAs per wave, K=32 in one shot
#pragma unroll
        for (int q = 0; q < 4; ++q)
            acc[q] = __builtin_amdgcn_wmma_f32_16x16x32_f16(
                false, ah.v, false, Bhh[q], (short)0, acc[q], false, false);

        // elementwise LSTM cell for this wave's column half; acc[0..3] = i,f,g,o tiles
#pragma unroll
        for (int r = 0; r < 8; ++r) {
            const float iv = fast_sig (acc[0][r]);
            const float fv = fast_sig (acc[1][r]);
            const float gv = fast_tanh(acc[2][r]);
            const float ov = fast_sig (acc[3][r]);
            const float cn = fmaf(fv, cs[r], iv * gv);
            cs[r] = cn;
            const float hv = ov * fast_tanh(cn);
            hbuf[(hi * 8 + r) * P + wv * 16 + n] = (_Float16)hv;
        }
        __syncthreads();                         // h_t fully visible to both waves

        if constexpr (MODE != 2) {
            // coalesced h_t store: 64 lanes cover 16 rows x 4 chunks of 16B
            const int m  = tid & 15;
            const int ch = tid >> 4;             // 0..3
            const uint4 qv = *(const uint4*)&hbuf[m * P + ch * 8];
            *(uint4*)(hout + ((size_t)(b0 + m) * Tn + t) * Hn + ch * 8) = qv;
        }
    }

    if constexpr (MODE == 2) {
#pragma unroll
        for (int r = 0; r < 8; ++r)
            c_out[(size_t)(b0 + hi * 8 + r) * Hn + wv * 16 + n] = cs[r];
    }
}

// mean / log_var heads: [512,32] x [32,16] each — trivial VALU kernel
__global__ void __launch_bounds__(256)
proj_kernel(const float* __restrict__ c3,
            const float* __restrict__ Wm, const float* __restrict__ bm,
            const float* __restrict__ Wl, const float* __restrict__ bl,
            float* __restrict__ out)
{
    const int idx = blockIdx.x * blockDim.x + threadIdx.x;
    if (idx >= Bn * Ln * 2) return;
    const int which = idx / (Bn * Ln);           // 0 = mean, 1 = log_var
    const int r     = idx - which * (Bn * Ln);
    const int b     = r >> 4;
    const int j     = r & 15;
    const float* W  = which ? Wl : Wm;
    float s         = which ? bl[j] : bm[j];
    const float* c  = c3 + (size_t)b * Hn;
    const float* w  = W  + (size_t)j * Hn;
#pragma unroll
    for (int k = 0; k < Hn; ++k) s = fmaf(c[k], w[k], s);
    out[idx] = s;                                // == out[which*8192 + b*16 + j]
}

extern "C" void kernel_launch(void* const* d_in, const int* in_sizes, int n_in,
                              void* d_out, int out_size, void* d_ws, size_t ws_size,
                              hipStream_t stream)
{
    const float* x      = (const float*)d_in[0];
    const float* W_ih0  = (const float*)d_in[1];
    const float* W_hh0  = (const float*)d_in[2];
    const float* b_ih0  = (const float*)d_in[3];
    const float* b_hh0  = (const float*)d_in[4];
    const float* W_ih1  = (const float*)d_in[5];
    const float* W_hh1  = (const float*)d_in[6];
    const float* b_ih1  = (const float*)d_in[7];
    const float* b_hh1  = (const float*)d_in[8];
    const float* W_ih2  = (const float*)d_in[9];
    const float* W_hh2  = (const float*)d_in[10];
    const float* b_ih2  = (const float*)d_in[11];
    const float* b_hh2  = (const float*)d_in[12];
    const float* W_mean = (const float*)d_in[13];
    const float* b_mean = (const float*)d_in[14];
    const float* W_logv = (const float*)d_in[15];
    const float* b_logv = (const float*)d_in[16];

    // workspace: two f16 h-sequences (32MB each) + final cell state (64KB)
    const size_t hseq_elems = (size_t)Bn * Tn * Hn;
    unsigned short* h0 = (unsigned short*)d_ws;
    unsigned short* h1 = h0 + hseq_elems;
    float*          c3 = (float*)(h1 + hseq_elems);

    const int tiles = Bn / 16;                   // 32 workgroups, 2 waves each

    lstm_wave_kernel<0><<<tiles, 64, 0, stream>>>(x,       nullptr, W_ih0, W_hh0, b_ih0, b_hh0, h0,      nullptr);
    lstm_wave_kernel<1><<<tiles, 64, 0, stream>>>(nullptr, h0,      W_ih1, W_hh1, b_ih1, b_hh1, h1,      nullptr);
    lstm_wave_kernel<2><<<tiles, 64, 0, stream>>>(nullptr, h1,      W_ih2, W_hh2, b_ih2, b_hh2, nullptr, c3);

    proj_kernel<<<(Bn * Ln * 2 + 255) / 256, 256, 0, stream>>>(c3, W_mean, b_mean, W_logv, b_logv, (float*)d_out);
}
